// DGM_77137612636306
// MI455X (gfx1250) — matrix operand order repaired and checked by reading, not compiled, not run on hardware
//
#include <hip/hip_runtime.h>
#include <hip/hip_bf16.h>

// ---------------------------------------------------------------------------
// MI455X (gfx1250) implementation.
//
// Key algebraic reduction: entmax15 support of row i is {j : vprobs_ij > T_i}
// and vprobs is an affine *decreasing* function of the distance D_ij per row
// (gamma=10>0), so res_ij = (D_ij < max(C_i, C_j)) with one per-row threshold
// C_i found by bisection on f(tau)=sum clip(x-tau)^2 - 1 (monotone, bracket
// [-1,0]).  This kills the reference's per-row 8192-element sort.
//
// The Gram matrix (8192x8192x64) is computed with v_wmma_f32_16x16x32_f16,
// using an f16 hi/lo split of x_emb (G ~= hh + hl + lh, fp32 accumulate) for
// near-fp32 distance accuracy.  D is never materialized: the only 256MB HBM
// traffic is the final `res` write (~11us at 23.3 TB/s), which is issued with
// non-temporal store hints so it streams past the 192MB L2 instead of
// evicting the ~2.4MB hot operand set (Xe hi/lo, aa, Cthr) that every tile
// re-reads.  Recomputing the Gram three times costs only ~77 GFLOP of f16
// WMMA — far below the WMMA roofline, so the whole problem is store-bound.
// ---------------------------------------------------------------------------

typedef _Float16 v16h __attribute__((ext_vector_type(16)));
typedef _Float16 v8h  __attribute__((ext_vector_type(8)));
typedef float    v8f  __attribute__((ext_vector_type(8)));

constexpr int N   = 8192;
constexpr int DIN = 128;
constexpr int DE  = 64;

__device__ __forceinline__ v8f wmma_f16(v16h a, v16h b, v8f c) {
  // v_wmma_f32_16x16x32_f16: (neg_a, A, neg_b, B, c_mod, C, reuse_a, reuse_b)
  return __builtin_amdgcn_wmma_f32_16x16x32_f16(false, a, false, b, (short)0, c,
                                                false, false);
}

// A-fragment (16x32 f16): lane<16 -> M=lane, K {0..15}+k0;
// lane>=16 -> M=lane-16, K {16..31}+k0.  Xe is row-major [N][64].
__device__ __forceinline__ v16h load_a_frag(const _Float16* Xe, int row0,
                                            int k0, int lane) {
  int m  = lane & 15;
  int kh = ((lane >> 4) << 4) + k0;
  return *(const v16h*)(Xe + (row0 + m) * DE + kh);
}

// B-fragment (32x16 f16): lane = column n (mod 16); lane<16 holds K=k0..k0+15,
// lane>=16 holds K=k0+16..k0+31 — contiguous in row-major Xe (no transpose).
__device__ __forceinline__ v16h load_b_frag(const _Float16* Xe, int col0,
                                            int k0, int lane) {
  int n  = lane & 15;
  int kh = ((lane >> 4) << 4) + k0;
  return *(const v16h*)(Xe + (col0 + n) * DE + kh);
}

struct AFrags { v16h h0, h1, l0, l1; };

__device__ __forceinline__ AFrags load_a(const _Float16* Xhi,
                                         const _Float16* Xlo,
                                         int row0, int lane) {
  AFrags A;
  A.h0 = load_a_frag(Xhi, row0, 0, lane);
  A.h1 = load_a_frag(Xhi, row0, 32, lane);
  A.l0 = load_a_frag(Xlo, row0, 0, lane);
  A.l1 = load_a_frag(Xlo, row0, 32, lane);
  return A;
}

// 16x16 Gram tile, error-compensated: G ~= lo*hi' + hi*lo' + hi*hi'
__device__ __forceinline__ v8f gram_tile(const AFrags& A, const _Float16* Xhi,
                                         const _Float16* Xlo, int col0,
                                         int lane) {
  v16h bh0 = load_b_frag(Xhi, col0, 0, lane);
  v16h bh1 = load_b_frag(Xhi, col0, 32, lane);
  v16h bl0 = load_b_frag(Xlo, col0, 0, lane);
  v16h bl1 = load_b_frag(Xlo, col0, 32, lane);
  v8f acc = {};
  acc = wmma_f16(A.l0, bh0, acc);
  acc = wmma_f16(A.l1, bh1, acc);
  acc = wmma_f16(A.h0, bl0, acc);
  acc = wmma_f16(A.h1, bl1, acc);
  acc = wmma_f16(A.h0, bh0, acc);
  acc = wmma_f16(A.h1, bh1, acc);
  return acc;
}

// ---------------------------------------------------------------------------
__global__ __launch_bounds__(256) void k_init(float* rowsum, float* rowsumsq,
                                              unsigned* rowmin, float* logp) {
  int i = blockIdx.x * blockDim.x + threadIdx.x;
  if (i < N) {
    rowsum[i]   = 0.f;
    rowsumsq[i] = 0.f;
    rowmin[i]   = 0x7F800000u;  // +inf
    logp[i]     = 0.f;          // atomically accumulated by k_adj
  }
}

// x_emb = x @ W (fp32), plus f16 hi/lo split for WMMA passes.
__global__ __launch_bounds__(256) void k_embed(const float* __restrict__ x,
                                               const float* __restrict__ W,
                                               float* xe, _Float16* Xhi,
                                               _Float16* Xlo) {
  int idx = blockIdx.x * 256 + threadIdx.x;  // idx = i*64 + e
  int i = idx >> 6, e = idx & 63;
  float acc = 0.f;
  #pragma unroll 8
  for (int k = 0; k < DIN; ++k)
    acc = fmaf(x[i * DIN + k], W[k * DE + e], acc);
  xe[idx] = acc;
  _Float16 hf = (_Float16)acc;
  Xhi[idx] = hf;
  Xlo[idx] = (_Float16)(acc - (float)hf);
}

__global__ __launch_bounds__(256) void k_aa(const float* __restrict__ xe,
                                            float* aa) {
  int i = blockIdx.x * blockDim.x + threadIdx.x;
  if (i >= N) return;
  float s = 0.f;
  #pragma unroll 8
  for (int e = 0; e < DE; ++e) {
    float v = xe[i * DE + e];
    s = fmaf(v, v, s);
  }
  aa[i] = s;
}

// Pass 1: per-row sum(D), sum(D^2), min off-diag D (layernorm stats + row max).
__global__ __launch_bounds__(256) void k_stats(const _Float16* __restrict__ Xhi,
                                               const _Float16* __restrict__ Xlo,
                                               const float* __restrict__ aa,
                                               float* rowsum, float* rowsumsq,
                                               unsigned* rowmin) {
  __shared__ float sSum[16], sSq[16];
  __shared__ unsigned sMin[16];
  int tid = threadIdx.x, lane = tid & 31, w = tid >> 5;
  if (tid < 16) { sSum[tid] = 0.f; sSq[tid] = 0.f; sMin[tid] = 0x7F800000u; }
  __syncthreads();

  int row0 = blockIdx.x * 16;
  AFrags A = load_a(Xhi, Xlo, row0, lane);
  int n = lane & 15, hw = lane >> 4;
  float aam[8];
  #pragma unroll
  for (int v = 0; v < 8; ++v) aam[v] = aa[row0 + hw * 8 + v];

  float sD[8], sQ[8], mn[8];
  #pragma unroll
  for (int v = 0; v < 8; ++v) { sD[v] = 0.f; sQ[v] = 0.f; mn[v] = __uint_as_float(0x7F800000u); }

  for (int t = 0; t < 4; ++t) {
    int j0 = ((blockIdx.y * 8 + w) * 4 + t) * 16;
    v8f g = gram_tile(A, Xhi, Xlo, j0, lane);
    int j = j0 + n;
    float aan = aa[j];
    #pragma unroll
    for (int v = 0; v < 8; ++v) {
      int i = row0 + hw * 8 + v;
      float d2 = fmaxf(aam[v] + aan - 2.f * g[v], 0.f);
      float d  = sqrtf(d2);
      if (i == j) { d = 0.f; d2 = 0.f; }        // diagonal distance is 0
      sD[v] += d;
      sQ[v] += d2;
      if (i != j) mn[v] = fminf(mn[v], d);
    }
  }
  #pragma unroll
  for (int off = 1; off < 16; off <<= 1) {
    #pragma unroll
    for (int v = 0; v < 8; ++v) {
      sD[v] += __shfl_xor(sD[v], off, 32);
      sQ[v] += __shfl_xor(sQ[v], off, 32);
      mn[v] = fminf(mn[v], __shfl_xor(mn[v], off, 32));
    }
  }
  if (n == 0) {
    #pragma unroll
    for (int v = 0; v < 8; ++v) {
      int r = hw * 8 + v;
      atomicAdd(&sSum[r], sD[v]);
      atomicAdd(&sSq[r], sQ[v]);
      atomicMin(&sMin[r], __float_as_uint(mn[v]));
    }
  }
  __syncthreads();
  if (tid < 16) {
    atomicAdd(&rowsum[row0 + tid], sSum[tid]);
    atomicAdd(&rowsumsq[row0 + tid], sSq[tid]);
    atomicMin(&rowmin[row0 + tid], sMin[tid]);
  }
}

// Per-row affine map:  x_ij (entmax score) = alpha_i * D_ij + beta_i  (j != i)
__global__ __launch_bounds__(256) void k_derive(const float* rowsum,
                                                const float* rowsumsq,
                                                const unsigned* rowmin,
                                                const float* gamma,
                                                const float* beta_in,
                                                float* alpha, float* beta_,
                                                float* Mrow) {
  int i = blockIdx.x * blockDim.x + threadIdx.x;
  if (i >= N) return;
  float g = gamma[0], b = beta_in[0];     // g = 10 (assumed > 0), b = 0
  float s = rowsum[i], q = rowsumsq[i];
  float meanD = s / (float)N;
  float var = fmaxf((q - s * meanD) / (float)(N - 1), 0.f);  // unbiased
  float denom = sqrtf(var) + 1e-6f;
  float meanP = -meanD;                                       // mean of probs
  float minD = __uint_as_float(rowmin[i]);
  // row max of vprobs over j!=i (vprobs decreasing in D), vs diag value -1e-6
  float vmax_off = g * (-minD - meanP) / denom + b;
  float M = 0.5f * fmaxf(vmax_off, -1e-6f);  // max of vprobs/2 (after diag fix)
  alpha[i] = -g / (2.f * denom);
  beta_[i] = 0.5f * (b - g * meanP / denom) - M;
  Mrow[i]  = M;
}

// Per-row entmax15 threshold via bisection over a 16-row f16 stripe in LDS.
__global__ __launch_bounds__(256) void k_tau(const _Float16* __restrict__ Xhi,
                                             const _Float16* __restrict__ Xlo,
                                             const float* __restrict__ aa,
                                             const float* __restrict__ alpha,
                                             const float* __restrict__ beta_,
                                             const float* __restrict__ Mrow,
                                             float* Cthr, float* dflag) {
  __shared__ __align__(16) _Float16 sRow[16 * N];  // 256 KB of the 320 KB WGP LDS
  int tid = threadIdx.x, lane = tid & 31, w = tid >> 5;
  int row0 = blockIdx.x * 16;
  AFrags A = load_a(Xhi, Xlo, row0, lane);
  int n = lane & 15, hw = lane >> 4;
  float aam[8], al[8], be[8];
  #pragma unroll
  for (int v = 0; v < 8; ++v) {
    int i = row0 + hw * 8 + v;
    aam[v] = aa[i]; al[v] = alpha[i]; be[v] = beta_[i];
  }
  // Phase 1: fill entmax scores x_ij for 16 rows x 8192 cols into LDS (f16).
  for (int t = 0; t < 64; ++t) {
    int j0 = (t * 8 + w) * 16;
    v8f g = gram_tile(A, Xhi, Xlo, j0, lane);
    int j = j0 + n;
    float aan = aa[j];
    #pragma unroll
    for (int v = 0; v < 8; ++v) {
      float d  = sqrtf(fmaxf(aam[v] + aan - 2.f * g[v], 0.f));
      float sv = fmaf(al[v], d, be[v]);
      sRow[(hw * 8 + v) * N + j] = (_Float16)sv;
    }
  }
  __syncthreads();
  if (tid < 16) {  // diagonal was replaced by -1e-6 *before* entmax
    int i = row0 + tid;
    sRow[tid * N + i] = (_Float16)(-5e-7f - Mrow[i]);
  }
  __syncthreads();

  // Phase 2: bisection; wave w owns rows w and w+8.  f(tau) monotone, root in
  // [-1, 0] since max(x)=0 and sum clip(x-tau)^2 = 1.
  for (int rr = 0; rr < 2; ++rr) {
    int r = w + rr * 8;
    const _Float16* row = &sRow[r * N];
    float lo = -1.f, hi = 0.f;
    for (int it = 0; it < 25; ++it) {
      float mid = 0.5f * (lo + hi);
      float part = 0.f;
      const v8h* p = (const v8h*)(row + lane * 256);
      #pragma unroll 4
      for (int c = 0; c < 32; ++c) {
        v8h x8 = p[c];
        #pragma unroll
        for (int e = 0; e < 8; ++e) {
          float xv = fmaxf((float)x8[e] - mid, 0.f);
          part = fmaf(xv, xv, part);
        }
      }
      #pragma unroll
      for (int off = 1; off < 32; off <<= 1) part += __shfl_xor(part, off, 32);
      bool ge = (part >= 1.f);
      lo = ge ? mid : lo;
      hi = ge ? hi : mid;
    }
    if (lane == 0) {
      float tau = 0.5f * (lo + hi);
      int i = row0 + r;
      Cthr[i]  = (tau - beta_[i]) / alpha[i];             // D_ij < C_i  <=>  in support
      dflag[i] = ((-5e-7f - Mrow[i]) > tau) ? 1.f : 0.f;  // diagonal membership
    }
  }
}

// Pass 3: res_ij = (D_ij < max(C_i, C_j)), diagonal overridden; row counts.
// res is written with non-temporal stores: 256MB of write-once data must not
// sweep the 192MB L2 and evict the cache-resident Gram operands.
__global__ __launch_bounds__(256) void k_adj(const _Float16* __restrict__ Xhi,
                                             const _Float16* __restrict__ Xlo,
                                             const float* __restrict__ aa,
                                             const float* __restrict__ Cthr,
                                             const float* __restrict__ dflag,
                                             float* __restrict__ res,
                                             float* __restrict__ logp) {
  __shared__ float sCnt[16];
  int tid = threadIdx.x, lane = tid & 31, w = tid >> 5;
  if (tid < 16) sCnt[tid] = 0.f;
  __syncthreads();

  int row0 = blockIdx.x * 16;
  AFrags A = load_a(Xhi, Xlo, row0, lane);
  int n = lane & 15, hw = lane >> 4;
  float aam[8], Cm[8], dfm[8], cnt[8];
  #pragma unroll
  for (int v = 0; v < 8; ++v) {
    int i = row0 + hw * 8 + v;
    aam[v] = aa[i]; Cm[v] = Cthr[i]; dfm[v] = dflag[i]; cnt[v] = 0.f;
  }
  for (int t = 0; t < 4; ++t) {
    int j0 = ((blockIdx.y * 8 + w) * 4 + t) * 16;
    v8f g = gram_tile(A, Xhi, Xlo, j0, lane);
    int j = j0 + n;
    float aan = aa[j], Cn = Cthr[j];
    #pragma unroll
    for (int v = 0; v < 8; ++v) {
      int i = row0 + hw * 8 + v;
      float d   = sqrtf(fmaxf(aam[v] + aan - 2.f * g[v], 0.f));
      float bit = (d < fmaxf(Cm[v], Cn)) ? 1.f : 0.f;
      if (i == j) bit = dfm[v];
      __builtin_nontemporal_store(bit, &res[(size_t)i * N + j]);
      cnt[v] += bit;
    }
  }
  #pragma unroll
  for (int off = 1; off < 16; off <<= 1)
    #pragma unroll
    for (int v = 0; v < 8; ++v) cnt[v] += __shfl_xor(cnt[v], off, 32);
  if (n == 0) {
    #pragma unroll
    for (int v = 0; v < 8; ++v) atomicAdd(&sCnt[hw * 8 + v], cnt[v]);
  }
  __syncthreads();
  if (tid < 16) atomicAdd(&logp[row0 + tid], sCnt[tid]);
}

// ---------------------------------------------------------------------------
extern "C" void kernel_launch(void* const* d_in, const int* in_sizes, int n_in,
                              void* d_out, int out_size, void* d_ws,
                              size_t ws_size, hipStream_t stream) {
  (void)in_sizes; (void)n_in; (void)out_size; (void)ws_size;
  const float* x     = (const float*)d_in[0];
  const float* W     = (const float*)d_in[1];
  const float* gamma = (const float*)d_in[2];
  const float* beta  = (const float*)d_in[3];
  // d_in[4] = edges (int64) — unused by the reference embed path.

  float* xe   = (float*)d_out;              // [N, DE]
  float* res  = xe + (size_t)N * DE;        // [N, N]
  float* logp = res + (size_t)N * N;        // [N]

  char* ws = (char*)d_ws;
  _Float16* Xhi = (_Float16*)ws;  ws += (size_t)N * DE * sizeof(_Float16);
  _Float16* Xlo = (_Float16*)ws;  ws += (size_t)N * DE * sizeof(_Float16);
  float*    aa       = (float*)ws;    ws += N * sizeof(float);
  float*    rowsum   = (float*)ws;    ws += N * sizeof(float);
  float*    rowsumsq = (float*)ws;    ws += N * sizeof(float);
  unsigned* rowmin   = (unsigned*)ws; ws += N * sizeof(unsigned);
  float*    alpha    = (float*)ws;    ws += N * sizeof(float);
  float*    beta_    = (float*)ws;    ws += N * sizeof(float);
  float*    Mrow     = (float*)ws;    ws += N * sizeof(float);
  float*    Cthr     = (float*)ws;    ws += N * sizeof(float);
  float*    dflag    = (float*)ws;    ws += N * sizeof(float);

  k_init  <<<N / 256, 256, 0, stream>>>(rowsum, rowsumsq, rowmin, logp);
  k_embed <<<(N * DE) / 256, 256, 0, stream>>>(x, W, xe, Xhi, Xlo);
  k_aa    <<<N / 256, 256, 0, stream>>>(xe, aa);
  k_stats <<<dim3(N / 16, 16), 256, 0, stream>>>(Xhi, Xlo, aa, rowsum,
                                                 rowsumsq, rowmin);
  k_derive<<<N / 256, 256, 0, stream>>>(rowsum, rowsumsq, rowmin, gamma, beta,
                                        alpha, beta_, Mrow);
  k_tau   <<<N / 16, 256, 0, stream>>>(Xhi, Xlo, aa, alpha, beta_, Mrow,
                                       Cthr, dflag);
  k_adj   <<<dim3(N / 16, 16), 256, 0, stream>>>(Xhi, Xlo, aa, Cthr, dflag,
                                                 res, logp);
}
